// AlignWaveletFusion_72335839199355
// MI455X (gfx1250) — matrix operand order repaired and checked by reading, not compiled, not run on hardware
//
#include <hip/hip_runtime.h>

typedef _Float16 v16h __attribute__((ext_vector_type(16)));
typedef _Float16 v8h  __attribute__((ext_vector_type(8)));
typedef float    v8f  __attribute__((ext_vector_type(8)));

#define HDIM 768
#define NBAT 8
#define SEQ  2048

// db4 analysis filters, reversed (H0R/H1R) and forward (G0F/G1F)
__constant__ float c_h0r[8] = { 0.2303778133088965f, 0.7148465705529157f, 0.6308807679298589f,
  -0.027983769416859854f, -0.18703481171909309f, 0.030841381835560764f, 0.0328830116668852f,
  -0.010597401785069032f };
__constant__ float c_h1r[8] = { -0.010597401785069032f, -0.0328830116668852f, 0.030841381835560764f,
  0.18703481171909309f, -0.027983769416859854f, -0.6308807679298589f, 0.7148465705529157f,
  -0.2303778133088965f };
__constant__ float c_g0[8] = { -0.010597401785069032f, 0.0328830116668852f, 0.030841381835560764f,
  -0.18703481171909309f, -0.027983769416859854f, 0.6308807679298589f, 0.7148465705529157f,
  0.2303778133088965f };
__constant__ float c_g1[8] = { -0.2303778133088965f, 0.7148465705529157f, -0.6308807679298589f,
  -0.027983769416859854f, 0.18703481171909309f, 0.030841381835560764f, -0.0328830116668852f,
  -0.010597401785069032f };

// ---------------- f32 -> f16 convert (weights) ----------------
__global__ void k_f32_to_f16(const float* __restrict__ in, _Float16* __restrict__ out, long n) {
  long i = (long)blockIdx.x * blockDim.x + threadIdx.x;
  if (i < n) out[i] = (_Float16)in[i];
}

// ---------------- one analysis level (zero-pad, stride 2) ----------------
// in layout (24, n_in, 768); level-1 reads x (B,S,3H) with (j,b) permute.
__global__ void k_dwt(const float* __restrict__ in, const float* __restrict__ x,
                      int n_in, int n_out,
                      float* __restrict__ lo32, float* __restrict__ hi32,
                      _Float16* __restrict__ lo16, _Float16* __restrict__ hi16) {
  long idx = (long)blockIdx.x * blockDim.x + threadIdx.x;
  long total = 24L * n_out * HDIM;
  if (idx >= total) return;
  int h = (int)(idx % HDIM);
  long t2 = idx / HDIM;
  int t = (int)(t2 % n_out);
  int r = (int)(t2 / n_out);
  float accL = 0.f, accH = 0.f;
  int s0 = 2 * t - 6;  // pad_left == 6 at every level here
#pragma unroll
  for (int k = 0; k < 8; ++k) {
    int s = s0 + k;
    if (s < 0 || s >= n_in) continue;
    float v;
    if (in) {
      v = in[((long)r * n_in + s) * HDIM + h];
    } else {
      int j = r >> 3, b = r & 7;
      v = x[((long)b * SEQ + s) * (3 * HDIM) + j * HDIM + h];
    }
    accL += v * c_h0r[k];
    accH += v * c_h1r[k];
  }
  long o = ((long)r * n_out + t) * HDIM + h;
  lo32[o] = accL;
  hi32[o] = accH;
  hi16[o] = (_Float16)accH;
  if (lo16) lo16[o] = (_Float16)accL;
}

// ---------------- gated GEMM: fused[m] = sum_j sigmoid(A_j W_mj^T) .* A_j --------------
// A16/A32: 3 slabs (j) of (Mtot x 768); W: f16, index (m*mStride + j)*H*H, row g, col h.
// out: 3 slabs (m) of (Mtot x 768).
// Block = 256 threads = 8 waves, computes a 128x128 tile; each wave owns 32(M)x64(N)
// via 2x4 wmma accumulators, K swept in steps of 32 (768 = 24 steps).
// Out-of-range M rows are CLAMPED for all loads (branch-free K-loop AND gate stage);
// their results are garbage but are discarded by the single guarded store at the end.
// Sigmoid uses v_rcp_f32 (1-ulp) instead of IEEE divide: the gate already passed
// through an f16 matmul, so the correctly-rounded div sequence is wasted VALU.
__global__ __launch_bounds__(256, 1)
void k_gate_gemm(const _Float16* __restrict__ A16, const float* __restrict__ A32,
                 const _Float16* __restrict__ W, float* __restrict__ out,
                 int Mtot, int mStride) {
  const int m = blockIdx.z;
  const int blockM = blockIdx.x * 128;
  const int blockN = blockIdx.y * 128;
  const int wave = threadIdx.x >> 5;
  const int lane = threadIdx.x & 31;
  const int wm = wave >> 1;      // 0..3 : 32-row strip
  const int wn = wave & 1;       // 0..1 : 64-col strip
  const int lq = lane & 15;      // lane % 16
  const int lh = lane >> 4;      // lane half (0/1)

  // A-fragment rows (clamped) per ISA wave32 16-bit A layout
  int arow[2];
#pragma unroll
  for (int im = 0; im < 2; ++im) {
    int row = blockM + wm * 32 + im * 16 + lq;
    arow[im] = row < Mtot ? row : (Mtot - 1);
  }
  // Gate/store rows per C-matrix layout: M = e + 8*lh (per im subtile)
  int crow[2][8];
#pragma unroll
  for (int im = 0; im < 2; ++im)
#pragma unroll
    for (int e = 0; e < 8; ++e) {
      int row = blockM + wm * 32 + im * 16 + lh * 8 + e;
      crow[im][e] = row < Mtot ? row : (Mtot - 1);  // clamped for gate loads
    }
  // B-fragment / output columns (always in range: HDIM is a multiple of 128)
  int bcol[4];
#pragma unroll
  for (int in_ = 0; in_ < 4; ++in_) bcol[in_] = blockN + wn * 64 + in_ * 16 + lq;

  float fin[2][4][8];
#pragma unroll
  for (int a = 0; a < 2; ++a)
#pragma unroll
    for (int b = 0; b < 4; ++b)
#pragma unroll
      for (int e = 0; e < 8; ++e) fin[a][b][e] = 0.f;

  for (int j = 0; j < 3; ++j) {
    const _Float16* Aj = A16 + (long)j * Mtot * HDIM;
    const float*    Fj = A32 + (long)j * Mtot * HDIM;
    const _Float16* Wj = W + ((long)m * mStride + j) * (long)HDIM * HDIM;

    // Hoisted per-row/col base pointers: K offset folds into load immediates.
    const _Float16* aptr[2];
#pragma unroll
    for (int im = 0; im < 2; ++im) aptr[im] = Aj + (long)arow[im] * HDIM + lh * 16;
    const _Float16* bptr[4];
#pragma unroll
    for (int in_ = 0; in_ < 4; ++in_) {
      bptr[in_] = Wj + (long)bcol[in_] * HDIM + lh * 16;
      __builtin_prefetch(bptr[in_], 0, 0);   // global_prefetch_b8: warm WGP$ for this j
    }

    v8f acc[2][4];
#pragma unroll
    for (int a = 0; a < 2; ++a)
#pragma unroll
      for (int b = 0; b < 4; ++b) acc[a][b] = (v8f){};

    for (int kk = 0; kk < HDIM; kk += 32) {
      // A fragments: 16x32 f16, lane holds 16 contiguous K = kb..kb+15
      union { v16h v; v8h h[2]; } afrag[2];
#pragma unroll
      for (int im = 0; im < 2; ++im) {
        const _Float16* p = aptr[im] + kk;
        afrag[im].h[0] = *(const v8h*)p;
        afrag[im].h[1] = *(const v8h*)(p + 8);
      }
      // B fragments: 32x16 f16; lane holds 16 contiguous K for its column
      v16h bfrag[4];
#pragma unroll
      for (int in_ = 0; in_ < 4; ++in_) bfrag[in_] = *(const v16h*)(bptr[in_] + kk);
#pragma unroll
      for (int im = 0; im < 2; ++im)
#pragma unroll
        for (int in_ = 0; in_ < 4; ++in_)
          acc[im][in_] = __builtin_amdgcn_wmma_f32_16x16x32_f16(
              false, afrag[im].v, false, bfrag[in_], (short)0, acc[im][in_],
              false, false);
    }

    // gate: fin += sigmoid(acc) * A32[crow][col]  -- branch-free, rcp-based sigmoid
#pragma unroll
    for (int im = 0; im < 2; ++im)
#pragma unroll
      for (int in_ = 0; in_ < 4; ++in_)
#pragma unroll
        for (int e = 0; e < 8; ++e) {
          float g = acc[im][in_][e];
          float aval = Fj[(long)crow[im][e] * HDIM + bcol[in_]];
          float sg = __builtin_amdgcn_rcpf(1.0f + __expf(-g));  // v_rcp_f32
          fin[im][in_][e] += aval * sg;
        }
  }

  // single guarded store stage
  float* Om = out + (long)m * Mtot * HDIM;
#pragma unroll
  for (int im = 0; im < 2; ++im)
#pragma unroll
    for (int in_ = 0; in_ < 4; ++in_)
#pragma unroll
      for (int e = 0; e < 8; ++e) {
        int row = blockM + wm * 32 + im * 16 + lh * 8 + e;
        if (row < Mtot) Om[(long)row * HDIM + bcol[in_]] = fin[im][in_][e];
      }
}

// ---------------- one synthesis level (lhs_dilation=2, pad 1) ----------------
// lo physical row length loPhys, used length n (== hi length); out length 2n-6.
__global__ void k_isfb(const float* __restrict__ lo, int loPhys,
                       const float* __restrict__ hi, int n,
                       float* __restrict__ out) {
  int n_out = 2 * n - 6;
  long idx = (long)blockIdx.x * blockDim.x + threadIdx.x;
  long total = 24L * n_out * HDIM;
  if (idx >= total) return;
  int h = (int)(idx % HDIM);
  long t2 = idx / HDIM;
  int o = (int)(t2 % n_out);
  int r = (int)(t2 / n_out);
  float acc = 0.f;
#pragma unroll
  for (int k = 0; k < 8; ++k) {
    int d = o + k - 1;
    if (d < 0 || (d & 1)) continue;
    int i = d >> 1;
    if (i >= n) continue;
    acc += lo[((long)r * loPhys + i) * HDIM + h] * c_g0[k];
    acc += hi[((long)r * n + i) * HDIM + h] * c_g1[k];
  }
  out[((long)r * n_out + o) * HDIM + h] = acc;
}

// ---------------- final: (recon - x_perm) * weight[m] ----------------
__global__ void k_final(const float* __restrict__ recon, const float* __restrict__ x,
                        const float* __restrict__ w, float* __restrict__ out) {
  long idx = (long)blockIdx.x * blockDim.x + threadIdx.x;
  long total = 3L * NBAT * SEQ * HDIM;
  if (idx >= total) return;
  int h = (int)(idx % HDIM);
  long t = idx / HDIM;
  int s = (int)(t % SEQ);
  long rb = t / SEQ;
  int b = (int)(rb & 7);
  int m = (int)(rb >> 3);
  float xv = x[((long)b * SEQ + s) * (3 * HDIM) + m * HDIM + h];
  out[idx] = (recon[idx] - xv) * w[m];
}

extern "C" void kernel_launch(void* const* d_in, const int* in_sizes, int n_in,
                              void* d_out, int out_size, void* d_ws, size_t ws_size,
                              hipStream_t stream) {
  const float* x      = (const float*)d_in[0];
  const float* low_w  = (const float*)d_in[1];
  const float* high_w = (const float*)d_in[2];
  const float* weight = (const float*)d_in[3];
  float* out = (float*)d_out;

  const int n1 = 1027, n2 = 517, n3 = 262, n4 = 134;
  const long U = 24L * HDIM;  // elements per unit of sequence length

  char* ws = (char*)d_ws;
  auto alloc = [&](size_t bytes) -> void* {
    void* p = (void*)ws;
    ws += (bytes + 255) & ~(size_t)255;
    return p;
  };
  auto cdiv = [](long a, long b) -> long { return (a + b - 1) / b; };

  float* lo1 = (float*)alloc(U * n1 * 4); float* hi1 = (float*)alloc(U * n1 * 4);
  float* lo2 = (float*)alloc(U * n2 * 4); float* hi2 = (float*)alloc(U * n2 * 4);
  float* lo3 = (float*)alloc(U * n3 * 4); float* hi3 = (float*)alloc(U * n3 * 4);
  float* lo4 = (float*)alloc(U * n4 * 4); float* hi4 = (float*)alloc(U * n4 * 4);
  float* flo = (float*)alloc(U * n4 * 4);
  float* fh1 = (float*)alloc(U * n1 * 4); float* fh2 = (float*)alloc(U * n2 * 4);
  float* fh3 = (float*)alloc(U * n3 * 4); float* fh4 = (float*)alloc(U * n4 * 4);
  float* tA  = (float*)alloc(U * 1028 * 4);
  float* tB  = (float*)alloc(U * 2048 * 4);
  _Float16* h16_1 = (_Float16*)alloc(U * n1 * 2);
  _Float16* h16_2 = (_Float16*)alloc(U * n2 * 2);
  _Float16* h16_3 = (_Float16*)alloc(U * n3 * 2);
  _Float16* h16_4 = (_Float16*)alloc(U * n4 * 2);
  _Float16* l16_4 = (_Float16*)alloc(U * n4 * 2);
  _Float16* wl16  = (_Float16*)alloc(9L  * HDIM * HDIM * 2);
  _Float16* wh16  = (_Float16*)alloc(36L * HDIM * HDIM * 2);

  // weights -> f16
  {
    long nl = 9L * HDIM * HDIM, nh = 36L * HDIM * HDIM;
    k_f32_to_f16<<<cdiv(nl, 256), 256, 0, stream>>>(low_w, wl16, nl);
    k_f32_to_f16<<<cdiv(nh, 256), 256, 0, stream>>>(high_w, wh16, nh);
  }

  // analysis filterbank, 4 levels
  {
    long t;
    t = 24L * n1 * HDIM;
    k_dwt<<<cdiv(t, 256), 256, 0, stream>>>(nullptr, x, SEQ, n1, lo1, hi1, nullptr, h16_1);
    t = 24L * n2 * HDIM;
    k_dwt<<<cdiv(t, 256), 256, 0, stream>>>(lo1, nullptr, n1, n2, lo2, hi2, nullptr, h16_2);
    t = 24L * n3 * HDIM;
    k_dwt<<<cdiv(t, 256), 256, 0, stream>>>(lo2, nullptr, n2, n3, lo3, hi3, nullptr, h16_3);
    t = 24L * n4 * HDIM;
    k_dwt<<<cdiv(t, 256), 256, 0, stream>>>(lo3, nullptr, n3, n4, lo4, hi4, l16_4, h16_4);
  }

  // gated WMMA fusion per band
  {
    dim3 blk(256, 1, 1);
    // low band: W index (m*3 + j)
    {
      int Mtot = NBAT * n4;
      dim3 grd((unsigned)cdiv(Mtot, 128), HDIM / 128, 3);
      k_gate_gemm<<<grd, blk, 0, stream>>>(l16_4, lo4, wl16, flo, Mtot, 3);
    }
    // high bands: W index ((m*4 + k)*3 + j) = base(k*3) + m*12 + j
    const _Float16* hA16[4] = { h16_1, h16_2, h16_3, h16_4 };
    const float*    hA32[4] = { hi1, hi2, hi3, hi4 };
    float*          hOut[4] = { fh1, fh2, fh3, fh4 };
    const int       hN[4]   = { n1, n2, n3, n4 };
    for (int k = 0; k < 4; ++k) {
      int Mtot = NBAT * hN[k];
      dim3 grd((unsigned)cdiv(Mtot, 128), HDIM / 128, 3);
      k_gate_gemm<<<grd, blk, 0, stream>>>(hA16[k], hA32[k],
                                           wh16 + (long)k * 3 * HDIM * HDIM,
                                           hOut[k], Mtot, 12);
    }
  }

  // synthesis filterbank (with trim handled by loPhys vs n)
  {
    long t;
    t = 24L * (2 * n4 - 6) * HDIM;  // -> 262
    k_isfb<<<cdiv(t, 256), 256, 0, stream>>>(flo, n4, fh4, n4, tA);
    t = 24L * (2 * n3 - 6) * HDIM;  // -> 518
    k_isfb<<<cdiv(t, 256), 256, 0, stream>>>(tA, 262, fh3, n3, tB);
    t = 24L * (2 * n2 - 6) * HDIM;  // -> 1028
    k_isfb<<<cdiv(t, 256), 256, 0, stream>>>(tB, 518, fh2, n2, tA);
    t = 24L * (2 * n1 - 6) * HDIM;  // -> 2048
    k_isfb<<<cdiv(t, 256), 256, 0, stream>>>(tA, 1028, fh1, n1, tB);
  }

  // final residual * weight
  {
    long t = 3L * NBAT * SEQ * HDIM;
    k_final<<<cdiv(t, 256), 256, 0, stream>>>(tB, x, weight, out);
  }
}
